// CMPNNEncoder_5583457485397
// MI455X (gfx1250) — matrix operand order repaired and checked by reading, not compiled, not run on hardware
//
#include <hip/hip_runtime.h>
#include <hip/hip_bf16.h>
#include <cstdint>
#include <cstddef>

#define N_NODES 32768
#define N_GRAPHS 512
#define SEQ 64
#define E_EDGES 131072
#define H 128
#define K_STEPS 5

typedef __attribute__((ext_vector_type(16))) __bf16 v16bf;
typedef __attribute__((ext_vector_type(8)))  __bf16 v8bf;
typedef __attribute__((ext_vector_type(8)))  float  v8f;

// ---------------------------------------------------------------- WMMA helpers

__device__ __forceinline__ v8f wmma_bf16(v16bf a, v16bf b, v8f c) {
  // D = A(16x32 bf16) * B(32x16 bf16) + C(16x16 f32)
  return __builtin_amdgcn_wmma_f32_16x16x32_bf16(false, a, false, b, (short)0, c,
                                                 false, false);
}

// A fragment: row-major A[M,Kp] bf16.  Per ISA 7.12.2 (16-bit A 16x32):
// lane L (L<16) holds M=row0+L, K = {k0..k0+7} in v0..3 and {k0+16..k0+23} in v4..7;
// lane L+16 holds same row, K offset +8.
__device__ __forceinline__ v16bf load_frag_a(const __bf16* A, int lda, int row0, int k0) {
  const int lane = threadIdx.x & 31;
  const int r    = row0 + (lane & 15);
  const int hi   = lane >> 4;
  const __bf16* p = A + (size_t)r * lda + k0 + hi * 8;
  union { v16bf v; v8bf h[2]; } u;
  u.h[0] = *(const v8bf*)(p);
  u.h[1] = *(const v8bf*)(p + 16);
  return u.v;
}

// B fragment from weight W[N,Kp] row-major (C = A * W^T): lane L (L<16) holds
// column n=col0+L with K = {k0..k0+7, k0+16..k0+23}; lane L+16 holds same column, K offset +8.
__device__ __forceinline__ v16bf load_frag_b(const __bf16* W, int ldw, int col0, int k0) {
  const int lane = threadIdx.x & 31;
  const int c    = col0 + (lane & 15);
  const int hi   = lane >> 4;
  const __bf16* p = W + (size_t)c * ldw + k0 + hi * 8;
  union { v16bf v; v8bf h[2]; } u;
  u.h[0] = *(const v8bf*)(p);
  u.h[1] = *(const v8bf*)(p + 16);
  return u.v;
}

// C/D 16x16 f32: VGPR i -> M = row0 + (lane>=16 ? 8 : 0) + i, N = col0 + (lane&15)
__device__ __forceinline__ void store_tile_f32(float* C, int ldc, int row0, int col0,
                                               v8f acc) {
  const int lane = threadIdx.x & 31;
  const int n  = col0 + (lane & 15);
  const int mb = row0 + ((lane >> 4) << 3);
#pragma unroll
  for (int i = 0; i < 8; ++i) C[(size_t)(mb + i) * ldc + n] = acc[i];
}

// ---------------------------------------------------------------- generic GEMM
// C[M,N] = epilogue(A[M,Kp] * W[N,Kp]^T + bias).
// One wave computes a 16x64 strip (4 accumulators): one A fragment is reused
// across 4 B fragments -> ~2.5 b128 loads per WMMA instead of 4, and 4 WMMAs
// in flight per A-load.  All operands stay L2-resident on MI455X (192MB L2),
// so fragments stream straight from global memory; no LDS staging needed.
// EPI 0: C = acc + bias.   EPI 1: C = relu(extra + acc + bias)
template <int EPI>
__global__ __launch_bounds__(256) void k_gemm(const __bf16* __restrict__ A,
                                              const __bf16* __restrict__ W,
                                              const float* __restrict__ bias,
                                              const float* __restrict__ extra,
                                              float* __restrict__ C,
                                              int M, int N, int Kp) {
  const int wave   = blockIdx.x * 8 + (threadIdx.x >> 5);
  const int strips = N >> 6;                 // N / 64
  const int tm = (wave / strips) << 4;
  const int tn = (wave % strips) << 6;
  if (tm >= M) return;
  v8f acc0 = {0.f, 0.f, 0.f, 0.f, 0.f, 0.f, 0.f, 0.f};
  v8f acc1 = acc0, acc2 = acc0, acc3 = acc0;
  for (int k0 = 0; k0 < Kp; k0 += 32) {
    v16bf a = load_frag_a(A, Kp, tm, k0);
    acc0 = wmma_bf16(a, load_frag_b(W, Kp, tn +  0, k0), acc0);
    acc1 = wmma_bf16(a, load_frag_b(W, Kp, tn + 16, k0), acc1);
    acc2 = wmma_bf16(a, load_frag_b(W, Kp, tn + 32, k0), acc2);
    acc3 = wmma_bf16(a, load_frag_b(W, Kp, tn + 48, k0), acc3);
  }
  const int lane = threadIdx.x & 31;
  const int mb = tm + ((lane >> 4) << 3);
  v8f accs[4] = {acc0, acc1, acc2, acc3};
#pragma unroll
  for (int j = 0; j < 4; ++j) {
    const int n = tn + j * 16 + (lane & 15);
    const float bi = bias ? bias[n] : 0.f;
#pragma unroll
    for (int i = 0; i < 8; ++i) {
      const size_t off = (size_t)(mb + i) * N + n;
      float v = accs[j][i] + bi;
      if (EPI == 1) v = fmaxf(0.f, extra[off] + v);
      C[off] = v;
    }
  }
}

// ---------------------------------------------------------------- elementwise

__global__ void k_fill(float* p, float v, long n) {
  long i = (long)blockIdx.x * blockDim.x + threadIdx.x;
  if (i < n) p[i] = v;
}

__global__ void k_cvt(const float* __restrict__ s, __bf16* __restrict__ d, long n) {
  long i = (long)blockIdx.x * blockDim.x + threadIdx.x;
  if (i < n) d[i] = (__bf16)s[i];
}

__global__ void k_cvt_pad(const float* __restrict__ s, __bf16* __restrict__ d,
                          int M, int K, int Kp) {
  long i = (long)blockIdx.x * blockDim.x + threadIdx.x;
  if (i >= (long)M * Kp) return;
  int m = (int)(i / Kp), k = (int)(i % Kp);
  d[i] = (k < K) ? (__bf16)s[(size_t)m * K + k] : (__bf16)0.f;
}

// LayerNorm over H=128, one wave per row (32 lanes x 4 floats)
// MODE 0: relu(LN(in))  1: LN(res + relu(in))  2: tanh(relu(LN(in)))  3: LN(in)
template <int MODE>
__global__ __launch_bounds__(256) void k_ln(const float* __restrict__ in,
                                            const float* __restrict__ res,
                                            const float* __restrict__ g,
                                            const float* __restrict__ b,
                                            float* __restrict__ out,
                                            float* __restrict__ out2,
                                            __bf16* __restrict__ outbf, int M) {
  const int row  = blockIdx.x * 8 + (threadIdx.x >> 5);
  const int lane = threadIdx.x & 31;
  if (row >= M) return;
  const size_t base = (size_t)row * H + lane * 4;
  float v[4];
#pragma unroll
  for (int i = 0; i < 4; ++i) v[i] = in[base + i];
  if (MODE == 1) {
#pragma unroll
    for (int i = 0; i < 4; ++i) v[i] = res[base + i] + fmaxf(0.f, v[i]);
  }
  float s = v[0] + v[1] + v[2] + v[3];
#pragma unroll
  for (int off = 16; off; off >>= 1) s += __shfl_xor(s, off);
  const float mean = s * (1.f / H);
  float ss = 0.f;
#pragma unroll
  for (int i = 0; i < 4; ++i) { float d = v[i] - mean; ss += d * d; }
#pragma unroll
  for (int off = 16; off; off >>= 1) ss += __shfl_xor(ss, off);
  const float inv = rsqrtf(ss * (1.f / H) + 1e-5f);
#pragma unroll
  for (int i = 0; i < 4; ++i) {
    const int c = lane * 4 + i;
    float y = (v[i] - mean) * inv * g[c] + b[c];
    if (MODE == 0) y = fmaxf(0.f, y);
    if (MODE == 2) y = tanhf(fmaxf(0.f, y));
    out[base + i] = y;
    if (out2)  out2[base + i]  = y;
    if (outbf) outbf[base + i] = (__bf16)y;
  }
}

// segment_sum (+ optional relu-segment_max via positive-only int atomicMax trick:
// relu(segment_max(x)) == max over {0} U {x}, and only positive candidates can
// win against a 0.0f-initialized buffer, where float ordering == int ordering)
template <bool WITHMAX>
__global__ void k_seg(const float* __restrict__ he, const int* __restrict__ dst,
                      float* __restrict__ msum, float* __restrict__ mmax) {
  long i = (long)blockIdx.x * blockDim.x + threadIdx.x;
  if (i >= (long)E_EDGES * H) return;
  const int e = (int)(i >> 7), f = (int)(i & 127);
  const float v = he[i];
  const size_t o = (size_t)dst[e] * H + f;
  __hip_atomic_fetch_add(&msum[o], v, __ATOMIC_RELAXED, __HIP_MEMORY_SCOPE_AGENT);
  if (WITHMAX && v > 0.f) atomicMax((int*)&mmax[o], __float_as_int(v));
}

__global__ void k_concat_msg(const float* __restrict__ hv, const float* __restrict__ ms,
                             const float* __restrict__ mm, __bf16* __restrict__ cat) {
  long i = (long)blockIdx.x * blockDim.x + threadIdx.x;
  if (i >= (long)N_NODES * H) return;
  const int nd = (int)(i >> 7), f = (int)(i & 127);
  cat[(size_t)nd * 256 + f]       = (__bf16)hv[i];
  cat[(size_t)nd * 256 + 128 + f] = (__bf16)(ms[i] * mm[i]);
}

__global__ void k_edge_gather(const float* __restrict__ hv, const float* __restrict__ he,
                              const int* __restrict__ dst, const int* __restrict__ inv,
                              __bf16* __restrict__ mebf) {
  long i = (long)blockIdx.x * blockDim.x + threadIdx.x;
  if (i >= (long)E_EDGES * H) return;
  const int e = (int)(i >> 7), f = (int)(i & 127);
  mebf[i] = (__bf16)(hv[(size_t)dst[e] * H + f] - he[(size_t)inv[e] * H + f]);
}

__global__ void k_boost(const float* __restrict__ hv, const float* __restrict__ bias,
                        __bf16* __restrict__ hb) {
  long i = (long)blockIdx.x * blockDim.x + threadIdx.x;
  if (i >= (long)N_NODES * H) return;
  hb[i] = (__bf16)fmaxf(0.f, hv[i] + bias[i & 127]);
}

__global__ void k_concat3(const float* __restrict__ a, const float* __restrict__ b,
                          const float* __restrict__ c, __bf16* __restrict__ cat) {
  long i = (long)blockIdx.x * blockDim.x + threadIdx.x;
  if (i >= (long)N_NODES * H) return;
  const int nd = (int)(i >> 7), f = (int)(i & 127);
  cat[(size_t)nd * 384 + f]       = (__bf16)a[i];
  cat[(size_t)nd * 384 + 128 + f] = (__bf16)b[i];
  cat[(size_t)nd * 384 + 256 + f] = (__bf16)c[i];
}

// ---------------------------------------------------------------- GRU (persistent)
// One block owns 16 graphs, loops all 64 steps internally (recurrence only
// couples time within a graph, so only block-level barriers are needed).
// Recurrent gh = h @ whh^T (16x128 * 128x384) via WMMA from LDS bf16 hidden
// state; whh stays L2-resident across all 64 steps.
__device__ __forceinline__ float sigm(float x) { return 1.f / (1.f + __expf(-x)); }

template <bool REVERSE, bool WRITE_SEQ>
__global__ __launch_bounds__(256) void k_bigru(const float* __restrict__ gi,
                                               const __bf16* __restrict__ whh,
                                               const float* __restrict__ bhh,
                                               __bf16* __restrict__ seq_out,
                                               float* __restrict__ z_out, int dircol) {
  __shared__ float  gh[16][384];
  __shared__ float  hf[16][H];
  __shared__ __bf16 hb[16][H];
  const int tid = threadIdx.x;
  const int g0  = blockIdx.x * 16;
  for (int i = tid; i < 16 * H; i += 256) { hf[i >> 7][i & 127] = 0.f; hb[i >> 7][i & 127] = (__bf16)0.f; }
  __syncthreads();
  const int w = tid >> 5;
  for (int s = 0; s < SEQ; ++s) {
    const int t = REVERSE ? (SEQ - 1 - s) : s;
    v8f a0 = {0.f,0.f,0.f,0.f,0.f,0.f,0.f,0.f};
    v8f a1 = a0, a2 = a0;
#pragma unroll
    for (int k0 = 0; k0 < H; k0 += 32) {
      v16bf af = load_frag_a(&hb[0][0], H, 0, k0);
      a0 = wmma_bf16(af, load_frag_b(whh, H, w * 48 +  0, k0), a0);
      a1 = wmma_bf16(af, load_frag_b(whh, H, w * 48 + 16, k0), a1);
      a2 = wmma_bf16(af, load_frag_b(whh, H, w * 48 + 32, k0), a2);
    }
    store_tile_f32(&gh[0][0], 384, 0, w * 48 +  0, a0);
    store_tile_f32(&gh[0][0], 384, 0, w * 48 + 16, a1);
    store_tile_f32(&gh[0][0], 384, 0, w * 48 + 32, a2);
    __syncthreads();
    for (int idx = tid; idx < 16 * H; idx += 256) {
      const int r = idx >> 7, f = idx & 127;
      const size_t node = (size_t)(g0 + r) * SEQ + t;
      const float* git = gi + node * 384;
      const float rr = sigm(git[f]       + gh[r][f]       + bhh[f]);
      const float zz = sigm(git[128 + f] + gh[r][128 + f] + bhh[128 + f]);
      const float nn = tanhf(git[256 + f] + rr * (gh[r][256 + f] + bhh[256 + f]));
      const float h  = (1.f - zz) * nn + zz * hf[r][f];
      hf[r][f] = h;
      if (WRITE_SEQ) {
        const float c = fminf(10.f, fmaxf(-10.f, h));
        seq_out[node * 256 + dircol + f] = (__bf16)c;
      }
    }
    __syncthreads();
    for (int idx = tid; idx < 16 * H; idx += 256) {
      const int r = idx >> 7, f = idx & 127;
      hb[r][f] = (__bf16)hf[r][f];
    }
    __syncthreads();
  }
  if (!WRITE_SEQ) {
    for (int idx = tid; idx < 16 * H; idx += 256) {
      const int r = idx >> 7, f = idx & 127;
      const float c = fminf(10.f, fmaxf(-10.f, hf[r][f]));
      z_out[(size_t)(g0 + r) * 256 + dircol + f] = c;
    }
  }
}

// ---------------------------------------------------------------- head

__global__ __launch_bounds__(256) void k_bn(const float* __restrict__ z,
                                            const float* __restrict__ g,
                                            const float* __restrict__ b,
                                            float* __restrict__ zn) {
  __shared__ float s1[256], s2[256];
  const int c = blockIdx.x, t = threadIdx.x;
  const float a0 = z[(size_t)t * 256 + c], a1 = z[(size_t)(t + 256) * 256 + c];
  s1[t] = a0 + a1; s2[t] = a0 * a0 + a1 * a1;
  __syncthreads();
  for (int off = 128; off; off >>= 1) {
    if (t < off) { s1[t] += s1[t + off]; s2[t] += s2[t + off]; }
    __syncthreads();
  }
  const float mu  = s1[0] * (1.f / N_GRAPHS);
  const float var = s2[0] * (1.f / N_GRAPHS) - mu * mu;
  const float inv = rsqrtf(var + 1e-5f);
  zn[(size_t)t * 256 + c]         = (a0 - mu) * inv * g[c] + b[c];
  zn[(size_t)(t + 256) * 256 + c] = (a1 - mu) * inv * g[c] + b[c];
}

__global__ __launch_bounds__(256) void k_pred(const float* __restrict__ zn,
                                              const float* __restrict__ pw,
                                              const float* __restrict__ pb,
                                              float* __restrict__ out) {
  const int w = threadIdx.x >> 5, lane = threadIdx.x & 31;
  const int gidx = blockIdx.x * 8 + w;
  float s = 0.f;
  for (int c = lane; c < 256; c += 32) s += zn[(size_t)gidx * 256 + c] * pw[c];
#pragma unroll
  for (int off = 16; off; off >>= 1) s += __shfl_xor(s, off);
  if (lane == 0) out[gidx] = s + pb[0];
}

// ---------------------------------------------------------------- host

static inline int cdiv(long a, long b) { return (int)((a + b - 1) / b); }

static void gemm(hipStream_t st, const __bf16* A, const __bf16* W, const float* bias,
                 const float* extra, float* C, int M, int N, int Kp, int epi) {
  const int blocks = ((M / 16) * (N / 64) + 7) / 8;
  if (epi == 0)
    k_gemm<0><<<blocks, 256, 0, st>>>(A, W, bias, nullptr, C, M, N, Kp);
  else
    k_gemm<1><<<blocks, 256, 0, st>>>(A, W, bias, extra, C, M, N, Kp);
}

extern "C" void kernel_launch(void* const* d_in, const int* in_sizes, int n_in,
                              void* d_out, int out_size, void* d_ws, size_t ws_size,
                              hipStream_t stream) {
  (void)in_sizes; (void)n_in; (void)out_size; (void)ws_size;
  // ---- inputs (setup_inputs order) ----
  const float* x         = (const float*)d_in[0];
  const float* edge_attr = (const float*)d_in[1];
  const float *node_w = (const float*)d_in[2],  *node_b  = (const float*)d_in[3];
  const float *node_g = (const float*)d_in[4],  *node_bt = (const float*)d_in[5];
  const float *edge_w = (const float*)d_in[6],  *edge_b  = (const float*)d_in[7];
  const float *edge_g = (const float*)d_in[8],  *edge_bt = (const float*)d_in[9];
  const float *lx_w   = (const float*)d_in[10], *lx_b    = (const float*)d_in[11];
  const float *lx_g   = (const float*)d_in[12], *lx_bt   = (const float*)d_in[13];
  const float *nu_w   = (const float*)d_in[14], *nu_b    = (const float*)d_in[15];
  const float *nn_g   = (const float*)d_in[16], *nn_b    = (const float*)d_in[17];
  const float *eu_w   = (const float*)d_in[18], *eu_b    = (const float*)d_in[19];
  const float *boost_bias = (const float*)d_in[20];
  const float *bg_wih_f = (const float*)d_in[21], *bg_whh_f = (const float*)d_in[22];
  const float *bg_bih_f = (const float*)d_in[23], *bg_bhh_f = (const float*)d_in[24];
  const float *bg_wih_b = (const float*)d_in[25], *bg_whh_b = (const float*)d_in[26];
  const float *bg_bih_b = (const float*)d_in[27], *bg_bhh_b = (const float*)d_in[28];
  const float *proj_w = (const float*)d_in[29], *proj_b  = (const float*)d_in[30];
  const float *proj_g = (const float*)d_in[31], *proj_bt = (const float*)d_in[32];
  const float *fin_w  = (const float*)d_in[33], *fin_b   = (const float*)d_in[34];
  const float *fin_g  = (const float*)d_in[35], *fin_bt  = (const float*)d_in[36];
  const float *rg_wih_f = (const float*)d_in[37], *rg_whh_f = (const float*)d_in[38];
  const float *rg_bih_f = (const float*)d_in[39], *rg_bhh_f = (const float*)d_in[40];
  const float *rg_wih_b = (const float*)d_in[41], *rg_whh_b = (const float*)d_in[42];
  const float *rg_bih_b = (const float*)d_in[43], *rg_bhh_b = (const float*)d_in[44];
  const float *bn_g = (const float*)d_in[45], *bn_b = (const float*)d_in[46];
  const float *pred_w = (const float*)d_in[47], *pred_b = (const float*)d_in[48];
  const int* edge_index = (const int*)d_in[49];
  const int* inv_idx    = (const int*)d_in[50];
  const int* dst        = edge_index + E_EDGES;  // row 1
  float* out = (float*)d_out;

  // ---- workspace bump allocator (deterministic) ----
  char* wsp = (char*)d_ws;
  auto alloc = [&](size_t bytes) -> char* {
    char* p = wsp; wsp += (bytes + 255) & ~(size_t)255; return p;
  };
  __bf16* XBF   = (__bf16*)alloc((size_t)N_NODES * 160 * 2);
  __bf16* EABF  = (__bf16*)alloc((size_t)E_EDGES * 32 * 2);
  __bf16* W_NODE = (__bf16*)alloc(128 * 160 * 2);
  __bf16* W_EDGE = (__bf16*)alloc(128 * 32 * 2);
  __bf16* W_LX   = (__bf16*)alloc(128 * 160 * 2);
  __bf16* W_NU   = (__bf16*)alloc((size_t)K_STEPS * 128 * 256 * 2);
  __bf16* W_EU   = (__bf16*)alloc((size_t)K_STEPS * 128 * 128 * 2);
  __bf16* W_BG_WIHF = (__bf16*)alloc(384 * 128 * 2);
  __bf16* W_BG_WHHF = (__bf16*)alloc(384 * 128 * 2);
  __bf16* W_BG_WIHB = (__bf16*)alloc(384 * 128 * 2);
  __bf16* W_BG_WHHB = (__bf16*)alloc(384 * 128 * 2);
  __bf16* W_RG_WIHF = (__bf16*)alloc(384 * 128 * 2);
  __bf16* W_RG_WHHF = (__bf16*)alloc(384 * 128 * 2);
  __bf16* W_RG_WIHB = (__bf16*)alloc(384 * 128 * 2);
  __bf16* W_RG_WHHB = (__bf16*)alloc(384 * 128 * 2);
  __bf16* W_PROJ = (__bf16*)alloc(128 * 256 * 2);
  __bf16* W_FIN  = (__bf16*)alloc(128 * 384 * 2);
  float*  HV   = (float*)alloc((size_t)N_NODES * H * 4);
  float*  HE0  = (float*)alloc((size_t)E_EDGES * H * 4);
  float*  HE   = (float*)alloc((size_t)E_EDGES * H * 4);
  char*   MEBF_raw = alloc((size_t)E_EDGES * H * 2);      // multi-use bf16 region
  float*  MSUM = (float*)alloc((size_t)N_NODES * H * 4);  // also m_final
  float*  MMAX = (float*)alloc((size_t)N_NODES * H * 4);  // also h_x
  __bf16* CAT  = (__bf16*)alloc((size_t)N_NODES * 256 * 2); // msg-cat / hb_bf / hv2_bf
  float*  T1   = (float*)alloc((size_t)N_NODES * H * 4);
  float*  GIF  = (float*)alloc((size_t)N_NODES * 384 * 4);
  float*  GIB  = (float*)alloc((size_t)N_NODES * 384 * 4);
  float*  Z    = (float*)alloc((size_t)N_GRAPHS * 256 * 4);
  float*  ZN   = (float*)alloc((size_t)N_GRAPHS * 256 * 4);
  __bf16* MEBF = (__bf16*)MEBF_raw;          // m_e bf16 [E,128]
  __bf16* SEQBF = (__bf16*)MEBF_raw;         // GRU seq out bf16 [N,256]
  __bf16* CAT3  = (__bf16*)MEBF_raw;         // final concat bf16 [N,384]
  float*  HV2   = HE0;                       // h_v2 fp32 (HE0 dead after loop)
  __bf16* HV2BF = CAT;
  __bf16* HBBF  = CAT;

  const long NH = (long)N_NODES * H, EH = (long)E_EDGES * H;

  // ---- convert / pad weights + inputs to bf16 ----
  k_cvt_pad<<<cdiv((long)N_NODES * 160, 256), 256, 0, stream>>>(x, XBF, N_NODES, 133, 160);
  k_cvt_pad<<<cdiv((long)E_EDGES * 32, 256), 256, 0, stream>>>(edge_attr, EABF, E_EDGES, 14, 32);
  k_cvt_pad<<<cdiv(128 * 160, 256), 256, 0, stream>>>(node_w, W_NODE, 128, 133, 160);
  k_cvt_pad<<<cdiv(128 * 32, 256), 256, 0, stream>>>(edge_w, W_EDGE, 128, 14, 32);
  k_cvt_pad<<<cdiv(128 * 160, 256), 256, 0, stream>>>(lx_w, W_LX, 128, 133, 160);
  k_cvt<<<cdiv((long)K_STEPS * 128 * 256, 256), 256, 0, stream>>>(nu_w, W_NU, (long)K_STEPS * 128 * 256);
  k_cvt<<<cdiv((long)K_STEPS * 128 * 128, 256), 256, 0, stream>>>(eu_w, W_EU, (long)K_STEPS * 128 * 128);
  k_cvt<<<cdiv(384 * 128, 256), 256, 0, stream>>>(bg_wih_f, W_BG_WIHF, 384 * 128);
  k_cvt<<<cdiv(384 * 128, 256), 256, 0, stream>>>(bg_whh_f, W_BG_WHHF, 384 * 128);
  k_cvt<<<cdiv(384 * 128, 256), 256, 0, stream>>>(bg_wih_b, W_BG_WIHB, 384 * 128);
  k_cvt<<<cdiv(384 * 128, 256), 256, 0, stream>>>(bg_whh_b, W_BG_WHHB, 384 * 128);
  k_cvt<<<cdiv(384 * 128, 256), 256, 0, stream>>>(rg_wih_f, W_RG_WIHF, 384 * 128);
  k_cvt<<<cdiv(384 * 128, 256), 256, 0, stream>>>(rg_whh_f, W_RG_WHHF, 384 * 128);
  k_cvt<<<cdiv(384 * 128, 256), 256, 0, stream>>>(rg_wih_b, W_RG_WIHB, 384 * 128);
  k_cvt<<<cdiv(384 * 128, 256), 256, 0, stream>>>(rg_whh_b, W_RG_WHHB, 384 * 128);
  k_cvt<<<cdiv(128 * 256, 256), 256, 0, stream>>>(proj_w, W_PROJ, 128 * 256);
  k_cvt<<<cdiv(128 * 384, 256), 256, 0, stream>>>(fin_w, W_FIN, 128 * 384);

  // ---- init node / edge embeddings ----
  gemm(stream, XBF, W_NODE, node_b, nullptr, T1, N_NODES, H, 160, 0);
  k_ln<0><<<N_NODES / 8, 256, 0, stream>>>(T1, nullptr, node_g, node_bt, HV, nullptr, nullptr, N_NODES);
  gemm(stream, EABF, W_EDGE, edge_b, nullptr, HE, E_EDGES, H, 32, 0);
  k_ln<0><<<E_EDGES / 8, 256, 0, stream>>>(HE, nullptr, edge_g, edge_bt, HE0, HE, nullptr, E_EDGES);

  // ---- K_STEPS of message passing ----
  for (int s = 0; s < K_STEPS; ++s) {
    k_fill<<<cdiv(NH, 256), 256, 0, stream>>>(MSUM, 0.f, NH);
    k_fill<<<cdiv(NH, 256), 256, 0, stream>>>(MMAX, 0.f, NH);
    k_seg<true><<<cdiv(EH, 256), 256, 0, stream>>>(HE, dst, MSUM, MMAX);
    k_concat_msg<<<cdiv(NH, 256), 256, 0, stream>>>(HV, MSUM, MMAX, CAT);
    gemm(stream, CAT, W_NU + (size_t)s * 128 * 256, nu_b + s * H, nullptr, T1, N_NODES, H, 256, 0);
    k_ln<1><<<N_NODES / 8, 256, 0, stream>>>(T1, HV, nn_g + s * H, nn_b + s * H, HV, nullptr, nullptr, N_NODES);
    k_edge_gather<<<cdiv(EH, 256), 256, 0, stream>>>(HV, HE, dst, inv_idx, MEBF);
    gemm(stream, MEBF, W_EU + (size_t)s * 128 * 128, eu_b + s * H, HE0, HE, E_EDGES, H, 128, 1);
  }

  // ---- BatchGRUBooster ----
  k_boost<<<cdiv(NH, 256), 256, 0, stream>>>(HV, boost_bias, HBBF);
  gemm(stream, HBBF, W_BG_WIHF, bg_bih_f, nullptr, GIF, N_NODES, 384, 128, 0);
  gemm(stream, HBBF, W_BG_WIHB, bg_bih_b, nullptr, GIB, N_NODES, 384, 128, 0);
  k_bigru<false, true><<<N_GRAPHS / 16, 256, 0, stream>>>(GIF, W_BG_WHHF, bg_bhh_f, SEQBF, nullptr, 0);
  k_bigru<true,  true><<<N_GRAPHS / 16, 256, 0, stream>>>(GIB, W_BG_WHHB, bg_bhh_b, SEQBF, nullptr, 128);
  gemm(stream, SEQBF, W_PROJ, proj_b, nullptr, T1, N_NODES, H, 256, 0);
  k_ln<2><<<N_NODES / 8, 256, 0, stream>>>(T1, nullptr, proj_g, proj_bt, HV, nullptr, nullptr, N_NODES);

  // ---- final merge ----
  k_fill<<<cdiv(NH, 256), 256, 0, stream>>>(MSUM, 0.f, NH);
  k_seg<false><<<cdiv(EH, 256), 256, 0, stream>>>(HE, dst, MSUM, nullptr);
  gemm(stream, XBF, W_LX, lx_b, nullptr, T1, N_NODES, H, 160, 0);
  k_ln<3><<<N_NODES / 8, 256, 0, stream>>>(T1, nullptr, lx_g, lx_bt, MMAX, nullptr, nullptr, N_NODES);
  k_concat3<<<cdiv(NH, 256), 256, 0, stream>>>(HV, MSUM, MMAX, CAT3);
  gemm(stream, CAT3, W_FIN, fin_b, nullptr, T1, N_NODES, H, 384, 0);
  k_ln<0><<<N_NODES / 8, 256, 0, stream>>>(T1, nullptr, fin_g, fin_bt, HV2, nullptr, HV2BF, N_NODES);

  // ---- readout BiGRU (final hidden states only) ----
  gemm(stream, HV2BF, W_RG_WIHF, rg_bih_f, nullptr, GIF, N_NODES, 384, 128, 0);
  gemm(stream, HV2BF, W_RG_WIHB, rg_bih_b, nullptr, GIB, N_NODES, 384, 128, 0);
  k_bigru<false, false><<<N_GRAPHS / 16, 256, 0, stream>>>(GIF, W_RG_WHHF, rg_bhh_f, nullptr, Z, 0);
  k_bigru<true,  false><<<N_GRAPHS / 16, 256, 0, stream>>>(GIB, W_RG_WHHB, rg_bhh_b, nullptr, Z, 128);

  // ---- BatchNorm (batch stats) + prediction head ----
  k_bn<<<256, 256, 0, stream>>>(Z, bn_g, bn_b, ZN);
  k_pred<<<N_GRAPHS / 8, 256, 0, stream>>>(ZN, pred_w, pred_b, out);
}
